// DepthVolumeModel_80736795230429
// MI455X (gfx1250) — matrix operand order repaired and checked by reading, not compiled, not run on hardware
//
// CDNA5 / gfx1250 implementation of the recurrent multi-view depth network.
//
// v2 changes vs v1 (from disasm feedback):
//  - A-fragment gather: per-element LDS offsets are now compile-time constants
//    (hi=0/1 variants selected by one cndmask) and UNCONDITIONAL (LDS tile padded
//    to CPAD channels, pad rows zero-filled) -> no per-element branch, no
//    per-element s_wait_dscnt, no div-by-9 magic sequences.
//  - Epilogue: no null-pointer checks, p1 gated by EPI at compile time,
//    contiguous unguarded stores (all W are exact multiples of WPB).
//  - All intermediate activations stored as f16 (bandwidth-shaped workload:
//    halves inter-kernel traffic); warp-source features kept channel-last so
//    each bilinear corner is 2x b128 loads instead of 16 strided scalars.
//  - Conv input staging for interior blocks uses the Tensor Data Mover
//    (tensor_load_to_lds, 3D tile cols x 3 rows x channels, OOB zero-fill =
//    free SAME padding right/bottom), wave0-issued + s_wait_tensorcnt(0);
//    boundary blocks use the cooperative path. __has_builtin-guarded.
//
// Input flattening assumed (setup_inputs insertion order, params recursed in
// insertion order): 0 src_images, 1 ys_dst, 2 xs_dst, 3 ys_src, 4 xs_src,
// 5 dst_K, 6 dst_E, 7 src_K, 8 src_E, 9..32 snet (w,b,g,be,m,v x4),
// 33..62 cell0..cell4 (wz,bz,wr,br,wq,bq), 63..70 deconv2/3 (w,b,g,be),
// 71 conv3_w, 72 conv3_b, 73 conv4_w, 74 conv4_b.  (75 inputs)

#include <hip/hip_runtime.h>

typedef __attribute__((ext_vector_type(16))) _Float16 v16h;
typedef __attribute__((ext_vector_type(8)))  _Float16 v8h;
typedef __attribute__((ext_vector_type(8)))  float    v8f;
typedef _Float16 h16;

#define NViews 4
#define IMG_H 256
#define IMG_W 256
#define IMG_HW (IMG_H * IMG_W)
#define NDEPTH 32

#if defined(__gfx1250__) && __has_builtin(__builtin_amdgcn_tensor_load_to_lds)
#define HAVE_TDM 1
typedef unsigned int v4u __attribute__((ext_vector_type(4)));
typedef int v4i __attribute__((ext_vector_type(4)));
typedef int v8i __attribute__((ext_vector_type(8)));
#else
#define HAVE_TDM 0
#endif

static __device__ __forceinline__ int iclamp(int v, int lo, int hi) {
  return v < lo ? lo : (v > hi ? hi : v);
}

// ---------------------------------------------------------------------------
// Weight packing: B operand for v_wmma_f32_16x16x32_f16, this file's own k order (A gather matches).
// B is 32x16 (KxN): lane L holds column n = L&15; lanes 0-15 hold K=0..15,
// lanes 16-31 K=16..31, K ascending within the lane's 16 halves.
// ---------------------------------------------------------------------------
__global__ void k_pack_w2(const float* __restrict__ wa, const float* __restrict__ wb,
                          h16* __restrict__ wp, int Cin, int coutA, int ksteps) {
  int idx = blockIdx.x * blockDim.x + threadIdx.x;
  int total = ksteps * 512;
  if (idx >= total) return;
  int j = idx & 15, lane = (idx >> 4) & 31, s = idx >> 9;
  int n = lane & 15;
  int k = s * 32 + ((lane >> 4) << 4) + j;
  int K = Cin * 9;
  int coutT = wb ? 2 * coutA : coutA;
  float v = 0.f;
  if (k < K && n < coutT) {
    const float* w = (n < coutA) ? wa : wb;
    int nn = (n < coutA) ? n : n - coutA;
    v = w[(size_t)nn * K + k];  // OIHW, k = cin*9 + ky*3 + kx
  }
  wp[(size_t)s * 512 + lane * 16 + j] = (h16)v;
}

__global__ void k_bn_fold(const float* b, const float* g, const float* be,
                          const float* m, const float* v, float* scale, float* shift, int C) {
  int i = threadIdx.x;
  if (i < C) {
    float s = g[i] * rsqrtf(v[i] + 1e-5f);
    scale[i] = s;
    shift[i] = (b[i] - m[i]) * s + be[i];
  }
}

__global__ void k_cat2(const float* a, const float* b, float* o, int c) {
  int i = threadIdx.x;
  if (i < c) { o[i] = a[i]; o[c + i] = b[i]; }
}

__global__ void k_cvt_h(const float* __restrict__ in, h16* __restrict__ out, int n) {
  int i = blockIdx.x * blockDim.x + threadIdx.x;
  if (i < n) out[i] = (h16)in[i];
}

// Planar (V,16,HW) f16 -> channel-last (V,HW,16) f16 for the warp sampler.
__global__ void k_feats_cl(const h16* __restrict__ in, h16* __restrict__ out) {
  int p = blockIdx.x * blockDim.x + threadIdx.x;
  if (p >= NViews * IMG_HW) return;
  int v = p / IMG_HW, q = p - v * IMG_HW;
  for (int c = 0; c < 16; ++c)
    out[((size_t)p << 4) + c] = in[((size_t)v * 16 + c) * IMG_HW + q];
}

#if HAVE_TDM
// TDM 3D tile load: cols (dim0) x 3 rows (dim1, stride rowstep) x nchan
// (dim2, stride chanstep), f16 elements, contiguous into LDS at lds_byte.
// OOB (cols_valid / rows_valid) reads return zero -> SAME padding for free.
static __device__ __forceinline__ void tdm_stage3(
    unsigned lds_byte, const h16* g, int cols_valid, int rows_valid,
    int nchan, int rowstep, long long chanstep, int tile_w) {
  unsigned long long ga = (unsigned long long)(size_t)g;
  v4u g0 = { 1u,                                    // count=1, user descriptor
             lds_byte,                               // lds_addr
             (unsigned)ga,                           // global_addr[31:0]
             (unsigned)((ga >> 32) & 0x01FFFFFFull) | 0x80000000u };  // addr[56:32] | type=2
  unsigned long long s0 = (unsigned long long)rowstep;
  unsigned long long s1 = (unsigned long long)chanstep;
  v8i g1 = {
    (int)0x00010000u,                                           // data_size=2B
    (int)(((unsigned)cols_valid & 0xFFFFu) << 16),              // tensor_dim0 lo16
    (int)((((unsigned)cols_valid >> 16) & 0xFFFFu) |
          (((unsigned)rows_valid & 0xFFFFu) << 16)),            // dim0 hi16 | dim1 lo16
    (int)(((unsigned)tile_w & 0xFFFFu) << 16),                  // dim1 hi16(0) | tile_dim0
    (int)(3u | ((unsigned)nchan << 16)),                        // tile_dim1=3 | tile_dim2
    (int)(unsigned)(s0 & 0xFFFFFFFFull),                        // dim0_stride[31:0]
    (int)(((unsigned)((s0 >> 32) & 0xFFFFull)) |
          (((unsigned)(s1 & 0xFFFFull)) << 16)),                // dim0_str[47:32] | dim1_str[15:0]
    (int)(unsigned)((s1 >> 16) & 0xFFFFFFFFull)                 // dim1_stride[47:16]
  };
  v4i g2 = { nchan, 0, 0, 0 };  // tensor_dim2; iterate off
  v4i g3 = { 0, 0, 0, 0 };
#if __clang_major__ >= 23
  v8i gz = { 0, 0, 0, 0, 0, 0, 0, 0 };
  __builtin_amdgcn_tensor_load_to_lds(g0, g1, g2, g3, gz, 0);
#else
  __builtin_amdgcn_tensor_load_to_lds(g0, g1, g2, g3, 0);
#endif
}
#endif

// ---------------------------------------------------------------------------
// Implicit-GEMM 3x3 conv on WMMA (f16 in, OT out).
//   KSTEPS: K-dim WMMA steps; WPB: pixels/block along a row; DIL: dilation;
//   EPI: 0=+bias, 1=scale/shift+relu, 2=sigmoid(+bias), 3=tanh(+bias),
//        4=scale/shift.
// LDS tile padded to CPAD channels so the A gather needs no bounds checks
// (pad rows zeroed; padded K has B==0).
// ---------------------------------------------------------------------------
template <int KSTEPS, int WPB, int DIL, int EPI, typename OT>
__global__ void k_conv3x3_wmma(const h16* __restrict__ in1, int cin1,
                               const h16* __restrict__ in2, int cin2,
                               const h16* __restrict__ in3, int cin3,
                               const h16* __restrict__ wp,
                               const float* __restrict__ p0,
                               const float* __restrict__ p1,
                               OT* __restrict__ out, int cout, int Cbuf,
                               int H, int W) {
  constexpr int CPAD = (KSTEPS * 32 - 1) / 9 + 1;  // max channel the gather touches
  constexpr int LW = WPB + 2 * DIL;
  __shared__ h16 lds[CPAD * 3 * LW];

  const int nv = blockIdx.z, y = blockIdx.y, xb = blockIdx.x * WPB;
  const int cinT = cin1 + cin2 + cin3;
  const size_t HW = (size_t)H * W;

  bool staged = false;
#if HAVE_TDM
  if (xb >= DIL && y >= DIL) {
    if (threadIdx.x < 32) {  // wave 0 issues the DMA
      const int x0 = xb - DIL, y0 = y - DIL;
      int rv = (H - 1 - y0) / DIL + 1;
      if (rv > 3) rv = 3;
      unsigned lb = (unsigned)(size_t)&lds[0];
      const h16* sA = in1 + (size_t)nv * cin1 * HW + (size_t)y0 * W + x0;
      tdm_stage3(lb, sA, W - x0, rv, cin1, DIL * W, (long long)HW, LW);
      if (cin2) {
        const h16* sB = in2 + (size_t)nv * cin2 * HW + (size_t)y0 * W + x0;
        tdm_stage3(lb + (unsigned)cin1 * 3u * (unsigned)LW * 2u, sB, W - x0, rv, cin2,
                   DIL * W, (long long)HW, LW);
      }
      if (cin3) {
        const h16* sC = in3 + (size_t)nv * cin3 * HW + (size_t)y0 * W + x0;
        tdm_stage3(lb + (unsigned)(cin1 + cin2) * 3u * (unsigned)LW * 2u, sC, W - x0, rv, cin3,
                   DIL * W, (long long)HW, LW);
      }
      __builtin_amdgcn_s_wait_tensorcnt(0);
    }
    // zero the channel pad region (read by the unconditional gather)
    for (int i = cinT * 3 * LW + threadIdx.x; i < CPAD * 3 * LW; i += blockDim.x)
      lds[i] = (h16)0.f;
    staged = true;
  }
#endif
  if (!staged) {
    for (int i = threadIdx.x; i < CPAD * 3 * LW; i += blockDim.x) {
      int col = i % LW;
      int t = i / LW;
      int r = t % 3, c = t / 3;
      int gx = xb - DIL + col;
      int gy = y + (r - 1) * DIL;
      float v = 0.f;
      if (c < cinT && gx >= 0 && gx < W && gy >= 0 && gy < H) {
        const h16* src;
        if (c < cin1)             src = in1 + ((size_t)nv * cin1 + c) * HW;
        else if (c < cin1 + cin2) src = in2 + ((size_t)nv * cin2 + (c - cin1)) * HW;
        else                      src = in3 + ((size_t)nv * cin3 + (c - cin1 - cin2)) * HW;
        v = (float)src[(size_t)gy * W + gx];
      }
      lds[i] = (h16)v;
    }
  }
  __syncthreads();

  const int lane = threadIdx.x & 31;
  const int wave = threadIdx.x >> 5;
  const int hi = lane >> 4, m = lane & 15;
  const int colbase = wave * 16 + m;  // + kx*DIL -> LDS column

  __builtin_prefetch(wp, 0, 0);

  v8f acc = {};
#pragma unroll
  for (int s = 0; s < KSTEPS; ++s) {
    v16h a;
#pragma unroll
    for (int j = 0; j < 16; ++j) {
      // A-fragment (16-bit A 16x32): k = hi*16 + j + s*32 (matches packed B).
      // Both hi variants fold to compile-time LDS offsets.
      const int k0 = s * 32 + j;
      const int k1 = k0 + 16;
      const int c0 = k0 / 9, q0i = k0 - c0 * 9;
      const int c1 = k1 / 9, q1i = k1 - c1 * 9;
      const int off0 = (c0 * 3 + q0i / 3) * LW + (q0i % 3) * DIL;
      const int off1 = (c1 * 3 + q1i / 3) * LW + (q1i % 3) * DIL;
      a[j] = lds[colbase + (hi ? off1 : off0)];
    }
    v16h b = *(const v16h*)(wp + (size_t)s * 512 + lane * 16);
    acc = __builtin_amdgcn_wmma_f32_16x16x32_f16(false, a, false, b,
                                                 (short)0, acc, false, false);
  }

  // D: lane holds column n = lane&15; rows m = r + 8*hi -> 8 contiguous pixels.
  const int n = m;
  if (n < cout) {
    const float q0 = p0[n];
    float q1 = 0.f;
    if (EPI == 1 || EPI == 4) q1 = p1[n];
    OT* op = out + ((size_t)nv * Cbuf + n) * HW + (size_t)y * W + (xb + wave * 16 + hi * 8);
#pragma unroll
    for (int r = 0; r < 8; ++r) {
      float v = acc[r];
      if (EPI == 0)      v += q0;
      else if (EPI == 1) { v = v * q0 + q1; v = v > 0.f ? v : 0.f; }
      else if (EPI == 2) { v = 1.f / (1.f + __expf(-(v + q0))); }
      else if (EPI == 3) { v = tanhf(v + q0); }
      else if (EPI == 4) { v = v * q0 + q1; }
      op[r] = (OT)v;
    }
  }
}

// ---------------------------------------------------------------------------
// Per-view projection setup: M = srcK * R * inv(dstK), t = srcK * T[:3,3],
// T = srcE * rigid_inv(dstE).
// ---------------------------------------------------------------------------
static __device__ void inv3(const float* K, float* o) {
  float a = K[0], b = K[1], c = K[2], d = K[3], e = K[4], f = K[5],
        g = K[6], h = K[7], i = K[8];
  float A = e * i - f * h, B = -(d * i - f * g), C = d * h - e * g;
  float id = 1.f / (a * A + b * B + c * C);
  o[0] = A * id;              o[1] = -(b * i - c * h) * id; o[2] = (b * f - c * e) * id;
  o[3] = B * id;              o[4] = (a * i - c * g) * id;  o[5] = -(a * f - c * d) * id;
  o[6] = C * id;              o[7] = -(a * h - b * g) * id; o[8] = (a * e - b * d) * id;
}

__global__ void k_proj(const float* dstK, const float* dstE, const float* srcK,
                       const float* srcE, float* Mv, float* tv) {
  int v = threadIdx.x;
  if (v >= NViews) return;
  const float* Kd = dstK + v * 9;
  const float* Ed = dstE + v * 16;
  const float* Ks = srcK + v * 9;
  const float* Es = srcE + v * 16;
  float iK[9];
  inv3(Kd, iK);
  float iR[9], it[3];
  for (int r = 0; r < 3; ++r)
    for (int c = 0; c < 3; ++c) iR[r * 3 + c] = Ed[c * 4 + r];
  for (int r = 0; r < 3; ++r)
    it[r] = -(iR[r * 3 + 0] * Ed[3] + iR[r * 3 + 1] * Ed[7] + iR[r * 3 + 2] * Ed[11]);
  float R[9], t3[3];
  for (int r = 0; r < 3; ++r) {
    for (int c = 0; c < 3; ++c) {
      float s = 0.f;
      for (int k = 0; k < 3; ++k) s += Es[r * 4 + k] * iR[k * 3 + c];
      R[r * 3 + c] = s;
    }
    float s = Es[r * 4 + 3];
    for (int k = 0; k < 3; ++k) s += Es[r * 4 + k] * it[k];
    t3[r] = s;
  }
  float RK[9];
  for (int r = 0; r < 3; ++r)
    for (int c = 0; c < 3; ++c) {
      float s = 0.f;
      for (int k = 0; k < 3; ++k) s += R[r * 3 + k] * iK[k * 3 + c];
      RK[r * 3 + c] = s;
    }
  for (int r = 0; r < 3; ++r) {
    for (int c = 0; c < 3; ++c) {
      float s = 0.f;
      for (int k = 0; k < 3; ++k) s += Ks[r * 3 + k] * RK[k * 3 + c];
      Mv[v * 9 + r * 3 + c] = s;
    }
    float s = 0.f;
    for (int k = 0; k < 3; ++k) s += Ks[r * 3 + k] * t3[k];
    tv[v * 3 + r] = s;
  }
}

// ---------------------------------------------------------------------------
// Fused warp + cost-volume + cell0-input assembly for one depth plane.
// featsCL is channel-last f16: each bilinear corner is 16 contiguous halves.
// ---------------------------------------------------------------------------
__global__ void k_warp_cost(const h16* __restrict__ featsCL, const float* __restrict__ Mv,
                            const float* __restrict__ tv,
                            const float* xsd, const float* ysd,
                            const float* xss, const float* yss,
                            float depth, h16* __restrict__ xout) {
  int p = blockIdx.x * blockDim.x + threadIdx.x;
  if (p >= IMG_HW) return;
  int py = p >> 8, px = p & 255;
  float f[NViews][16];
  for (int v = 0; v < NViews; ++v) {
    const float* M = Mv + v * 9;
    const float* t = tv + v * 3;
    float X = (float)px + xsd[v];
    float Y = (float)py + ysd[v];
    float r0 = M[0] * X + M[1] * Y + M[2];
    float r1 = M[3] * X + M[4] * Y + M[5];
    float r2 = M[6] * X + M[7] * Y + M[8];
    float p0 = r0 * depth + t[0];
    float p1 = r1 * depth + t[1];
    float p2 = r2 * depth + t[2];
    float z = (fabsf(p2) < 1e-6f) ? 1e-6f : p2;
    float u = p0 / z - xss[v];
    float w = p1 / z - yss[v];
    float x0 = floorf(u), y0 = floorf(w);
    float wx = u - x0, wy = w - y0;
    int xi0 = (int)x0, yi0 = (int)y0;
    int cx[2] = {xi0, xi0 + 1}, cy[2] = {yi0, yi0 + 1};
    float wxv[2] = {1.f - wx, wx}, wyv[2] = {1.f - wy, wy};
    for (int c = 0; c < 16; ++c) f[v][c] = 0.f;
    for (int jy = 0; jy < 2; ++jy)
      for (int jx = 0; jx < 2; ++jx) {
        int xi = cx[jx], yi = cy[jy];
        float valid = (xi >= 0 && xi <= IMG_W - 1 && yi >= 0 && yi <= IMG_H - 1) ? 1.f : 0.f;
        float wgt = wxv[jx] * wyv[jy] * valid;
        if (wgt != 0.f) {
          const h16* base = featsCL +
              (((size_t)v * IMG_HW + (size_t)iclamp(yi, 0, IMG_H - 1) * IMG_W +
                iclamp(xi, 0, IMG_W - 1)) << 4);
#pragma unroll
          for (int c = 0; c < 16; ++c) f[v][c] += wgt * (float)base[c];
        }
      }
  }
  float vc[NViews], vcm = 0.f;
  for (int mm = 0; mm < NViews; ++mm) {
    float s = 0.f;
    for (int v = 0; v < NViews; ++v) {
      float d = 0.f;
      for (int c = 0; c < 16; ++c) d += f[v][c] * f[mm][c];
      s += d;
    }
    vc[mm] = s * 0.25f;
    vcm += vc[mm];
  }
  vcm *= 0.25f;
  for (int i = 0; i < NViews; ++i) {
    h16* xo = xout + (size_t)i * 18 * IMG_HW + p;
    for (int c = 0; c < 16; ++c) xo[(size_t)c * IMG_HW] = (h16)f[i][c];
    xo[(size_t)16 * IMG_HW] = (h16)vc[i];
    xo[(size_t)17 * IMG_HW] = (h16)vcm;
  }
}

// ---------------------------------------------------------------------------
// GRU elementwise pieces (f16 storage, f32 math).
// ---------------------------------------------------------------------------
__global__ void k_rh(const h16* __restrict__ zr, const h16* __restrict__ h,
                     h16* __restrict__ rh, int cout, int HWp, int total) {
  int i = blockIdx.x * blockDim.x + threadIdx.x;
  if (i >= total) return;
  int hw = i % HWp;
  int t = i / HWp;
  int c = t % cout, nv = t / cout;
  rh[i] = (h16)((float)zr[((size_t)nv * 2 * cout + cout + c) * HWp + hw] * (float)h[i]);
}

__global__ void k_gru_up(const h16* __restrict__ zr, const h16* __restrict__ q,
                         h16* __restrict__ h, int cout, int HWp, int total) {
  int i = blockIdx.x * blockDim.x + threadIdx.x;
  if (i >= total) return;
  int hw = i % HWp;
  int t = i / HWp;
  int c = t % cout, nv = t / cout;
  float z = (float)zr[((size_t)nv * 2 * cout + c) * HWp + hw];
  h[i] = (h16)((1.f - z) * (float)h[i] + z * (float)q[i]);
}

__global__ void k_maxpool2(const h16* __restrict__ in, h16* __restrict__ out,
                           int CC, int H, int W) {
  int Ho = H / 2, Wo = W / 2;
  int total = CC * Ho * Wo;
  int i = blockIdx.x * blockDim.x + threadIdx.x;
  if (i >= total) return;
  int ox = i % Wo;
  int oy = (i / Wo) % Ho;
  int c = i / (Wo * Ho);
  const h16* b = in + ((size_t)c * H + oy * 2) * W + ox * 2;
  float v = fmaxf(fmaxf((float)b[0], (float)b[1]), fmaxf((float)b[W], (float)b[W + 1]));
  out[i] = (h16)v;
}

// conv_transpose stride-2, 16x16 kernel, SAME (pad 7). Direct accumulation.
__global__ void k_deconv(const h16* __restrict__ in, const float* __restrict__ w,
                         const float* __restrict__ b, float* __restrict__ out,
                         int Cin, int Cout, int Hin, int Win) {
  int Ho = Hin * 2, Wo = Win * 2;
  int p = blockIdx.x * blockDim.x + threadIdx.x;
  if (p >= Ho * Wo) return;
  int oy = p / Wo, ox = p % Wo;
  int co = blockIdx.y, nv = blockIdx.z;
  float acc = b[co];
  for (int ky = 0; ky < 16; ++ky) {
    int ty = oy + 7 - ky;
    if (ty & 1) continue;
    int iy = ty >> 1;
    if (iy < 0 || iy >= Hin) continue;
    for (int kx = 0; kx < 16; ++kx) {
      int tx = ox + 7 - kx;
      if (tx & 1) continue;
      int ix = tx >> 1;
      if (ix < 0 || ix >= Win) continue;
      for (int ci = 0; ci < Cin; ++ci)
        acc += (float)in[((size_t)(nv * Cin + ci) * Hin + iy) * Win + ix] *
               w[((size_t)(co * Cin + ci) * 16 + ky) * 16 + kx];
    }
  }
  out[((size_t)(nv * Cout + co) * Ho + oy) * Wo + ox] = acc;
}

// Instance-norm over (C,H,W) per image + affine + relu; f32 in, f16 out.
__global__ void k_inorm_relu(const float* __restrict__ in, const float* g,
                             const float* be, h16* __restrict__ out,
                             int C, int H, int W) {
  int nv = blockIdx.x;
  int n = C * H * W;
  const float* src = in + (size_t)nv * n;
  h16* dst = out + (size_t)nv * n;
  __shared__ float sb0[256], sb1[256];
  float s = 0.f, ss = 0.f;
  for (int i = threadIdx.x; i < n; i += 256) {
    float v = src[i];
    s += v;
    ss += v * v;
  }
  sb0[threadIdx.x] = s;
  sb1[threadIdx.x] = ss;
  __syncthreads();
  for (int o = 128; o > 0; o >>= 1) {
    if ((int)threadIdx.x < o) {
      sb0[threadIdx.x] += sb0[threadIdx.x + o];
      sb1[threadIdx.x] += sb1[threadIdx.x + o];
    }
    __syncthreads();
  }
  float mu = sb0[0] / n;
  float var = sb1[0] / n - mu * mu;
  float inv = rsqrtf(var + 1e-5f);
  for (int i = threadIdx.x; i < n; i += 256) {
    int c = i / (H * W);
    float v = (src[i] - mu) * inv * g[c] + be[c];
    dst[i] = (h16)(v > 0.f ? v : 0.f);
  }
}

// src_weight slice -> f32 output, mean over views of f3 ch1..8 -> cell4 input.
__global__ void k_f3_post(const h16* __restrict__ f3, h16* __restrict__ m4,
                          float* __restrict__ sw, int d) {
  int p = blockIdx.x * blockDim.x + threadIdx.x;
  if (p >= IMG_HW) return;
  for (int v = 0; v < NViews; ++v)
    sw[((size_t)v * NDEPTH + d) * IMG_HW + p] = (float)f3[(size_t)v * 9 * IMG_HW + p];
  for (int c = 1; c < 9; ++c) {
    float s = 0.f;
    for (int v = 0; v < NViews; ++v) s += (float)f3[((size_t)v * 9 + c) * IMG_HW + p];
    m4[(size_t)(c - 1) * IMG_HW + p] = (h16)(s * 0.25f);
  }
}

// ---------------------------------------------------------------------------
// Host orchestration.
// ---------------------------------------------------------------------------
static inline int cdiv_i(int a, int b) { return (a + b - 1) / b; }

extern "C" void kernel_launch(void* const* d_in, const int* in_sizes, int n_in,
                              void* d_out, int out_size, void* d_ws, size_t ws_size,
                              hipStream_t stream) {
  (void)in_sizes; (void)n_in; (void)out_size; (void)ws_size;
  const int H = IMG_H, W = IMG_W, HW = IMG_HW;

  // ---- inputs ----
  const float* src_images = (const float*)d_in[0];
  const float* ysd = (const float*)d_in[1];
  const float* xsd = (const float*)d_in[2];
  const float* yss = (const float*)d_in[3];
  const float* xss = (const float*)d_in[4];
  const float* dstK = (const float*)d_in[5];
  const float* dstE = (const float*)d_in[6];
  const float* srcK = (const float*)d_in[7];
  const float* srcE = (const float*)d_in[8];
  const float *SNW[4], *SNB[4], *SNG[4], *SNBE[4], *SNM[4], *SNV[4];
  for (int i = 0; i < 4; ++i) {
    int b = 9 + i * 6;
    SNW[i] = (const float*)d_in[b + 0]; SNB[i]  = (const float*)d_in[b + 1];
    SNG[i] = (const float*)d_in[b + 2]; SNBE[i] = (const float*)d_in[b + 3];
    SNM[i] = (const float*)d_in[b + 4]; SNV[i]  = (const float*)d_in[b + 5];
  }
  struct GruP { const float *wz, *bz, *wr, *br, *wq, *bq; } cell[5];
  for (int c = 0; c < 5; ++c) {
    int b = 33 + c * 6;
    cell[c].wz = (const float*)d_in[b + 0]; cell[c].bz = (const float*)d_in[b + 1];
    cell[c].wr = (const float*)d_in[b + 2]; cell[c].br = (const float*)d_in[b + 3];
    cell[c].wq = (const float*)d_in[b + 4]; cell[c].bq = (const float*)d_in[b + 5];
  }
  const float* dc2w = (const float*)d_in[63]; const float* dc2b = (const float*)d_in[64];
  const float* dc2g = (const float*)d_in[65]; const float* dc2be = (const float*)d_in[66];
  const float* dc3w = (const float*)d_in[67]; const float* dc3b = (const float*)d_in[68];
  const float* dc3g = (const float*)d_in[69]; const float* dc3be = (const float*)d_in[70];
  const float* c3w = (const float*)d_in[71]; const float* c3b = (const float*)d_in[72];
  const float* c4w = (const float*)d_in[73]; const float* c4b = (const float*)d_in[74];

  // ---- outputs: depth_probs (1,32,1,256,256) then src_weights (1,4,32,256,256)
  float* dp_out = (float*)d_out;
  float* sw_out = dp_out + (size_t)NDEPTH * HW;

  // ---- workspace carve (byte allocator, 64B aligned) ----
  char* base = (char*)d_ws;
  size_t off = 0;
  auto AB = [&](size_t bytes) {
    void* p = base + off;
    off = (off + bytes + 63) & ~(size_t)63;
    return p;
  };
  auto Ahf = [&](size_t n) { return (h16*)AB(n * 2); };
  auto Aff = [&](size_t n) { return (float*)AB(n * 4); };

  h16* img16  = Ahf((size_t)4 * 3 * HW);
  h16* feats  = Ahf((size_t)4 * 16 * HW);
  h16* featsCL= Ahf((size_t)4 * 16 * HW);
  h16* x18    = Ahf((size_t)4 * 18 * HW);
  h16* t1     = Ahf((size_t)4 * 8 * HW);
  h16* t2     = Ahf((size_t)4 * 8 * HW);
  h16* s0  = Ahf((size_t)4 * 8 * HW);
  h16* zr0 = Ahf((size_t)4 * 16 * HW);
  h16* rh0 = Ahf((size_t)4 * 8 * HW);
  h16* q0  = Ahf((size_t)4 * 8 * HW);
  h16* p0b = Ahf((size_t)4 * 8 * HW / 4);
  h16* s1  = Ahf((size_t)4 * 4 * HW / 4);
  h16* zr1 = Ahf((size_t)4 * 8 * HW / 4);
  h16* rh1 = Ahf((size_t)4 * 4 * HW / 4);
  h16* q1  = Ahf((size_t)4 * 4 * HW / 4);
  h16* p1b = Ahf((size_t)4 * 4 * HW / 16);
  h16* s2  = Ahf((size_t)4 * 4 * HW / 16);
  h16* zr2 = Ahf((size_t)4 * 8 * HW / 16);
  h16* rh2 = Ahf((size_t)4 * 4 * HW / 16);
  h16* q2  = Ahf((size_t)4 * 4 * HW / 16);
  float* d2 = Aff((size_t)4 * 3 * HW / 4);
  h16* d2n  = Ahf((size_t)4 * 3 * HW / 4);
  h16* s3  = Ahf((size_t)4 * 4 * HW / 4);
  h16* zr3 = Ahf((size_t)4 * 8 * HW / 4);
  h16* rh3 = Ahf((size_t)4 * 4 * HW / 4);
  h16* q3  = Ahf((size_t)4 * 4 * HW / 4);
  float* d3 = Aff((size_t)4 * 3 * HW);
  h16* d3n  = Ahf((size_t)4 * 3 * HW);
  h16* f3   = Ahf((size_t)4 * 9 * HW);
  h16* m4   = Ahf((size_t)8 * HW);
  h16* s4   = Ahf((size_t)4 * HW);
  h16* zr4  = Ahf((size_t)8 * HW);
  h16* rh4  = Ahf((size_t)4 * HW);
  h16* q4   = Ahf((size_t)4 * HW);
  float* Mv = Aff(NViews * 9);
  float* tv = Aff(NViews * 3);
  float* bnS[4]; float* bnH[4];
  for (int i = 0; i < 4; ++i) { bnS[i] = Aff(16); bnH[i] = Aff(16); }
  float* bzr[5];
  for (int c = 0; c < 5; ++c) bzr[c] = Aff(16);
  h16* wpS[4]  = {Ahf(1 * 512), Ahf(3 * 512), Ahf(3 * 512), Ahf(3 * 512)};
  h16* wpZR[5] = {Ahf(8 * 512), Ahf(4 * 512), Ahf(3 * 512), Ahf(4 * 512), Ahf(4 * 512)};
  h16* wpQ[5]  = {Ahf(8 * 512), Ahf(4 * 512), Ahf(3 * 512), Ahf(4 * 512), Ahf(4 * 512)};
  h16* wpC3 = Ahf(4 * 512);
  h16* wpC4 = Ahf(2 * 512);

  // ---- setup ----
  const int snC[4] = {8, 8, 8, 16};
  for (int i = 0; i < 4; ++i)
    k_bn_fold<<<1, 32, 0, stream>>>(SNB[i], SNG[i], SNBE[i], SNM[i], SNV[i], bnS[i], bnH[i], snC[i]);
  const int gcout[5] = {8, 4, 4, 4, 4};
  const int gcin[5]  = {18 + 8, 8 + 4, 4 + 4, 7 + 4, 8 + 4};
  const int gks[5]   = {8, 4, 3, 4, 4};
  for (int c = 0; c < 5; ++c) {
    k_cat2<<<1, 32, 0, stream>>>(cell[c].bz, cell[c].br, bzr[c], gcout[c]);
    k_pack_w2<<<cdiv_i(gks[c] * 512, 256), 256, 0, stream>>>(cell[c].wz, cell[c].wr, wpZR[c], gcin[c], gcout[c], gks[c]);
    k_pack_w2<<<cdiv_i(gks[c] * 512, 256), 256, 0, stream>>>(cell[c].wq, nullptr, wpQ[c], gcin[c], gcout[c], gks[c]);
  }
  const int snCin[4] = {3, 8, 8, 8};
  const int snKs[4] = {1, 3, 3, 3};
  for (int i = 0; i < 4; ++i)
    k_pack_w2<<<cdiv_i(snKs[i] * 512, 256), 256, 0, stream>>>(SNW[i], nullptr, wpS[i], snCin[i], snC[i], snKs[i]);
  k_pack_w2<<<cdiv_i(4 * 512, 256), 256, 0, stream>>>(c3w, nullptr, wpC3, 11, 9, 4);
  k_pack_w2<<<cdiv_i(2 * 512, 256), 256, 0, stream>>>(c4w, nullptr, wpC4, 4, 1, 2);
  k_proj<<<1, 32, 0, stream>>>(dstK, dstE, srcK, srcE, Mv, tv);
  k_cvt_h<<<cdiv_i(4 * 3 * HW, 256), 256, 0, stream>>>(src_images, img16, 4 * 3 * HW);

  // ---- zero recurrent state (graph-capture safe) ----
  hipMemsetAsync(s0, 0, (size_t)4 * 8 * HW * 2, stream);
  hipMemsetAsync(s1, 0, (size_t)4 * 4 * HW / 4 * 2, stream);
  hipMemsetAsync(s2, 0, (size_t)4 * 4 * HW / 16 * 2, stream);
  hipMemsetAsync(s3, 0, (size_t)4 * 4 * HW / 4 * 2, stream);
  hipMemsetAsync(s4, 0, (size_t)4 * HW * 2, stream);

  // ---- SNet feature extraction ----
  dim3 gFull(2, H, 4), gHalf(1, H / 2, 4), gQuar(1, H / 4, 4), gN1(2, H, 1);
  k_conv3x3_wmma<1, 128, 1, 1, h16><<<gFull, 256, 0, stream>>>(img16, 3, nullptr, 0, nullptr, 0, wpS[0], bnS[0], bnH[0], t1, 8, 8, H, W);
  k_conv3x3_wmma<3, 128, 1, 1, h16><<<gFull, 256, 0, stream>>>(t1, 8, nullptr, 0, nullptr, 0, wpS[1], bnS[1], bnH[1], t2, 8, 8, H, W);
  k_conv3x3_wmma<3, 128, 2, 1, h16><<<gFull, 256, 0, stream>>>(t2, 8, nullptr, 0, nullptr, 0, wpS[2], bnS[2], bnH[2], t1, 8, 8, H, W);
  k_conv3x3_wmma<3, 128, 1, 4, h16><<<gFull, 256, 0, stream>>>(t1, 8, nullptr, 0, nullptr, 0, wpS[3], bnS[3], bnH[3], feats, 16, 16, H, W);
  k_feats_cl<<<cdiv_i(4 * HW, 256), 256, 0, stream>>>(feats, featsCL);

  // ---- depth sweep ----
  for (int d = 0; d < NDEPTH; ++d) {
    float depth = 0.5f + (9.5f / 31.f) * d;
    k_warp_cost<<<HW / 256, 256, 0, stream>>>(featsCL, Mv, tv, xsd, ysd, xss, yss, depth, x18);

    // cell0 @256^2: cin 18+8=26, cout 8
    k_conv3x3_wmma<8, 128, 1, 2, h16><<<gFull, 256, 0, stream>>>(x18, 18, s0, 8, nullptr, 0, wpZR[0], bzr[0], bzr[0], zr0, 16, 16, H, W);
    k_rh<<<cdiv_i(4 * 8 * HW, 256), 256, 0, stream>>>(zr0, s0, rh0, 8, HW, 4 * 8 * HW);
    k_conv3x3_wmma<8, 128, 1, 3, h16><<<gFull, 256, 0, stream>>>(x18, 18, rh0, 8, nullptr, 0, wpQ[0], cell[0].bq, cell[0].bq, q0, 8, 8, H, W);
    k_gru_up<<<cdiv_i(4 * 8 * HW, 256), 256, 0, stream>>>(zr0, q0, s0, 8, HW, 4 * 8 * HW);

    // cell1 @128^2
    k_maxpool2<<<cdiv_i(4 * 8 * HW / 4, 256), 256, 0, stream>>>(s0, p0b, 32, H, W);
    k_conv3x3_wmma<4, 128, 1, 2, h16><<<gHalf, 256, 0, stream>>>(p0b, 8, s1, 4, nullptr, 0, wpZR[1], bzr[1], bzr[1], zr1, 8, 8, H / 2, W / 2);
    k_rh<<<cdiv_i(4 * 4 * HW / 4, 256), 256, 0, stream>>>(zr1, s1, rh1, 4, HW / 4, 4 * 4 * HW / 4);
    k_conv3x3_wmma<4, 128, 1, 3, h16><<<gHalf, 256, 0, stream>>>(p0b, 8, rh1, 4, nullptr, 0, wpQ[1], cell[1].bq, cell[1].bq, q1, 4, 4, H / 2, W / 2);
    k_gru_up<<<cdiv_i(4 * 4 * HW / 4, 256), 256, 0, stream>>>(zr1, q1, s1, 4, HW / 4, 4 * 4 * HW / 4);

    // cell2 @64^2
    k_maxpool2<<<cdiv_i(4 * 4 * HW / 16, 256), 256, 0, stream>>>(s1, p1b, 16, H / 2, W / 2);
    k_conv3x3_wmma<3, 64, 1, 2, h16><<<gQuar, 128, 0, stream>>>(p1b, 4, s2, 4, nullptr, 0, wpZR[2], bzr[2], bzr[2], zr2, 8, 8, H / 4, W / 4);
    k_rh<<<cdiv_i(4 * 4 * HW / 16, 256), 256, 0, stream>>>(zr2, s2, rh2, 4, HW / 16, 4 * 4 * HW / 16);
    k_conv3x3_wmma<3, 64, 1, 3, h16><<<gQuar, 128, 0, stream>>>(p1b, 4, rh2, 4, nullptr, 0, wpQ[2], cell[2].bq, cell[2].bq, q2, 4, 4, H / 4, W / 4);
    k_gru_up<<<cdiv_i(4 * 4 * HW / 16, 256), 256, 0, stream>>>(zr2, q2, s2, 4, HW / 16, 4 * 4 * HW / 16);

    // deconv2 + inst-norm + relu -> 3ch @128^2
    k_deconv<<<dim3(cdiv_i(HW / 4, 256), 3, 4), 256, 0, stream>>>(s2, dc2w, dc2b, d2, 4, 3, H / 4, W / 4);
    k_inorm_relu<<<4, 256, 0, stream>>>(d2, dc2g, dc2be, d2n, 3, H / 2, W / 2);

    // cell3 @128^2: [deconv(3) | s1(4)] + state s3(4) = 11
    k_conv3x3_wmma<4, 128, 1, 2, h16><<<gHalf, 256, 0, stream>>>(d2n, 3, s1, 4, s3, 4, wpZR[3], bzr[3], bzr[3], zr3, 8, 8, H / 2, W / 2);
    k_rh<<<cdiv_i(4 * 4 * HW / 4, 256), 256, 0, stream>>>(zr3, s3, rh3, 4, HW / 4, 4 * 4 * HW / 4);
    k_conv3x3_wmma<4, 128, 1, 3, h16><<<gHalf, 256, 0, stream>>>(d2n, 3, s1, 4, rh3, 4, wpQ[3], cell[3].bq, cell[3].bq, q3, 4, 4, H / 2, W / 2);
    k_gru_up<<<cdiv_i(4 * 4 * HW / 4, 256), 256, 0, stream>>>(zr3, q3, s3, 4, HW / 4, 4 * 4 * HW / 4);

    // deconv3 + inst-norm + relu -> 3ch @256^2 ; conv3 on [d3n(3)|s0(8)] -> 9ch
    k_deconv<<<dim3(cdiv_i(HW, 256), 3, 4), 256, 0, stream>>>(s3, dc3w, dc3b, d3, 4, 3, H / 2, W / 2);
    k_inorm_relu<<<4, 256, 0, stream>>>(d3, dc3g, dc3be, d3n, 3, H, W);
    k_conv3x3_wmma<4, 128, 1, 0, h16><<<gFull, 256, 0, stream>>>(d3n, 3, s0, 8, nullptr, 0, wpC3, c3b, c3b, f3, 9, 9, H, W);
    k_f3_post<<<HW / 256, 256, 0, stream>>>(f3, m4, sw_out, d);

    // cell4 @256^2, N=1
    k_conv3x3_wmma<4, 128, 1, 2, h16><<<gN1, 256, 0, stream>>>(m4, 8, s4, 4, nullptr, 0, wpZR[4], bzr[4], bzr[4], zr4, 8, 8, H, W);
    k_rh<<<cdiv_i(4 * HW, 256), 256, 0, stream>>>(zr4, s4, rh4, 4, HW, 4 * HW);
    k_conv3x3_wmma<4, 128, 1, 3, h16><<<gN1, 256, 0, stream>>>(m4, 8, rh4, 4, nullptr, 0, wpQ[4], cell[4].bq, cell[4].bq, q4, 4, 4, H, W);
    k_gru_up<<<cdiv_i(4 * HW, 256), 256, 0, stream>>>(zr4, q4, s4, 4, HW, 4 * HW);

    // conv4 -> depth prob slice d (f32 output)
    k_conv3x3_wmma<2, 128, 1, 0, float><<<gN1, 256, 0, stream>>>(s4, 4, nullptr, 0, nullptr, 0, wpC4, c4b, c4b, dp_out + (size_t)d * HW, 1, 1, H, W);
  }
}
